// MHSA_2808908611584
// MI455X (gfx1250) — matrix operand order repaired and checked by reading, not compiled, not run on hardware
//
#include <hip/hip_runtime.h>
#include <hip/hip_bf16.h>

// ---- vector types matching CDNA5 WMMA operand shapes (wave32) ----
typedef __attribute__((ext_vector_type(16))) __bf16 v16bf;
typedef __attribute__((ext_vector_type(8)))  __bf16 v8bf;
typedef __attribute__((ext_vector_type(8)))  float  v8f;

// Problem constants (fixed by reference)
#define BATCH 4
#define SEQ   2048
#define EMBED 1024
#define NHEAD 16
#define HDIM  64
#define F3    3072
#define MROWS (BATCH*SEQ)   // 8192

static __device__ __forceinline__ v8f vzero8() {
  v8f z;
#pragma unroll
  for (int i = 0; i < 8; ++i) z[i] = 0.0f;
  return z;
}

static __device__ __forceinline__ v8f wmma_bf16(v16bf a, v16bf b, v8f c) {
  // D = A(16x32 bf16) x B(32x16 bf16) + C(16x16 f32)
  return __builtin_amdgcn_wmma_f32_16x16x32_bf16(
      /*neg_a=*/false, a, /*neg_b=*/false, b,
      /*c_mod=*/(short)0, c, /*reuse_a=*/false, /*reuse_b=*/false);
}

// A-fragment (16x32 bf16). p = rowBase + k0 + hi*16 (32B aligned).
// halves [0..15] = K hi*16+[0..15].
static __device__ __forceinline__ v16bf a_frag_bf16(const __bf16* p) {
  return *(const v16bf*)p;
}

// B-fragment (32x16 bf16): lane = column, 16 contiguous K at hi*16.
// p = colBase + k0 + hi*16 (32B aligned).
static __device__ __forceinline__ v16bf b_frag_bf16(const __bf16* p) {
  return *(const v16bf*)p;
}

// Wave-level 16(M)x64(N) x K=1024 GEMM core, K unrolled 2x with ping-pong
// buffers: no rotation copies, one-iteration prefetch distance.
static __device__ __forceinline__ void gemm_core_16x64(
    const __bf16* __restrict__ arow, const __bf16* const bcol[4], v8f acc[4]) {
  v16bf a0 = a_frag_bf16(arow + 0);
  v16bf a1 = a_frag_bf16(arow + 32);
  v16bf b0[4], b1[4];
#pragma unroll
  for (int t = 0; t < 4; ++t) {
    b0[t] = b_frag_bf16(bcol[t] + 0);
    b1[t] = b_frag_bf16(bcol[t] + 32);
  }
  for (int k0 = 0; k0 < EMBED - 64; k0 += 64) {
#pragma unroll
    for (int t = 0; t < 4; ++t) acc[t] = wmma_bf16(a0, b0[t], acc[t]);
    a0 = a_frag_bf16(arow + k0 + 64);
#pragma unroll
    for (int t = 0; t < 4; ++t) b0[t] = b_frag_bf16(bcol[t] + k0 + 64);
#pragma unroll
    for (int t = 0; t < 4; ++t) acc[t] = wmma_bf16(a1, b1[t], acc[t]);
    a1 = a_frag_bf16(arow + k0 + 96);
#pragma unroll
    for (int t = 0; t < 4; ++t) b1[t] = b_frag_bf16(bcol[t] + k0 + 96);
  }
#pragma unroll
  for (int t = 0; t < 4; ++t) acc[t] = wmma_bf16(a0, b0[t], acc[t]);
#pragma unroll
  for (int t = 0; t < 4; ++t) acc[t] = wmma_bf16(a1, b1[t], acc[t]);
}

// ---------------------------------------------------------------------------
// Kernel 0: bulk f32 -> bf16 conversion (bandwidth-trivial pre-pass).
// ---------------------------------------------------------------------------
__global__ __launch_bounds__(256)
void cvt_f32_bf16_kernel(const float* __restrict__ src, __bf16* __restrict__ dst,
                         int n) {
  const int i = (blockIdx.x * 256 + threadIdx.x) * 8;
  if (i + 8 <= n) {
    float4 a = *(const float4*)(src + i);
    float4 b = *(const float4*)(src + i + 4);
    v8bf o;
    o[0] = (__bf16)a.x; o[1] = (__bf16)a.y; o[2] = (__bf16)a.z; o[3] = (__bf16)a.w;
    o[4] = (__bf16)b.x; o[5] = (__bf16)b.y; o[6] = (__bf16)b.z; o[7] = (__bf16)b.w;
    *(v8bf*)(dst + i) = o;
  }
}

// ---------------------------------------------------------------------------
// Kernel 1: qkv = x @ w_qkv^T + b_qkv (both operands pre-converted bf16).
// Emits Q (pre-scaled) / K as [B,H,N,D] bf16 and V transposed [B,H,D,N].
// Block = 256 thr (8 waves), workgroup tile 64(M) x 128(N), wave = 16x64.
// ---------------------------------------------------------------------------
__global__ __launch_bounds__(256)
void qkv_proj_kernel(const __bf16* __restrict__ x, const __bf16* __restrict__ w,
                     const float* __restrict__ bias,
                     __bf16* __restrict__ Qs, __bf16* __restrict__ Ks,
                     __bf16* __restrict__ Vt) {
  const int lane = threadIdx.x & 31;
  const int wv   = threadIdx.x >> 5;
  const int row  = lane & 15;
  const int hi   = lane >> 4;
  const int tileM = blockIdx.y * 64 + (wv & 3) * 16;
  const int tileN = blockIdx.x * 128 + (wv >> 2) * 64;

  const __bf16* arow = x + (size_t)(tileM + row) * EMBED + hi * 16;
  const __bf16* bcol[4];
#pragma unroll
  for (int t = 0; t < 4; ++t)
    bcol[t] = w + (size_t)(tileN + t * 16 + row) * EMBED + hi * 16;

  v8f acc[4];
#pragma unroll
  for (int t = 0; t < 4; ++t) acc[t] = vzero8();

  gemm_core_16x64(arow, bcol, acc);

#pragma unroll
  for (int t = 0; t < 4; ++t) {
    const int col0  = tileN + t * 16;
    const int which = col0 >> 10;   // 0=Q, 1=K, 2=V (uniform per 16-col tile)
#pragma unroll
    for (int i = 0; i < 8; ++i) {
      const int m   = tileM + hi * 8 + i;     // C layout: VGPR i -> row i / i+8
      const int col = col0 + row;             // lane&15 -> column
      float v = acc[t][i] + bias[col];
      const int hd = col & 1023, h = hd >> 6, d = hd & 63;
      const int n = m & (SEQ - 1), b = m >> 11;
      const size_t bh = (size_t)(b * NHEAD + h);
      if (which == 0)
        Qs[(bh * SEQ + n) * HDIM + d] = (__bf16)(v * 0.125f);  // 64^-0.5
      else if (which == 1)
        Ks[(bh * SEQ + n) * HDIM + d] = (__bf16)v;
      else
        Vt[(bh * HDIM + d) * SEQ + n] = (__bf16)v;             // transposed
    }
  }
}

// ---------------------------------------------------------------------------
// Kernel 2: streaming (flash) attention with cooperative LDS staging.
// 8 waves/block, wave owns 16 q rows. Per 32-key tile the block stages
// K (32x64) and V^T (64x32) into double-buffered LDS (shared by all 8
// waves -> 8x less L2 traffic); each wave: S = Q*K^T (4 wmma via ds_load
// fragments), online softmax (shfl_xor), P transpose through per-wave
// LDS, O += P*V (4 wmma). Next tile's global loads in flight during
// compute; ds_store + one barrier per iteration.
// ---------------------------------------------------------------------------
__global__ __launch_bounds__(256)
void attn_kernel(const __bf16* __restrict__ Qs, const __bf16* __restrict__ Ks,
                 const __bf16* __restrict__ Vt, __bf16* __restrict__ O) {
  __shared__ alignas(64) __bf16 kt[2][32 * HDIM];   // [key][d]   4 KB x2
  __shared__ alignas(64) __bf16 vt[2][HDIM * 32];   // [d][key]   4 KB x2
  __shared__ alignas(64) __bf16 plds[8][16 * 32];   // per-wave P tile

  const int tid  = threadIdx.x;
  const int lane = tid & 31;
  const int wv   = tid >> 5;
  const int row  = lane & 15;
  const int hi   = lane >> 4;

  const int bh = blockIdx.y;            // 0..63
  const int b  = bh >> 4, h = bh & 15;
  const int q0 = blockIdx.x * 128 + wv * 16;

  const __bf16* Qb = Qs + (size_t)bh * SEQ * HDIM;
  const __bf16* Kb = Ks + (size_t)bh * SEQ * HDIM;
  const __bf16* Vb = Vt + (size_t)bh * HDIM * SEQ;

  // staging assignments (256 threads, 2048 elems per tile, 8 elems each)
  const int vrow = tid >> 2;            // 0..63  (d row of V^T)
  const int voff = (tid & 3) * 8;       // 0/8/16/24 within 32-key row

  // Q fragments (K-dim = d = 64 -> two 16x32 A-frags), loaded once.
  const v16bf aq0 = a_frag_bf16(Qb + (size_t)(q0 + row) * HDIM + 0  + hi * 16);
  const v16bf aq1 = a_frag_bf16(Qb + (size_t)(q0 + row) * HDIM + 32 + hi * 16);

  v8f o[4];
  float mrow[8], lrow[8];
#pragma unroll
  for (int t = 0; t < 4; ++t) o[t] = vzero8();
#pragma unroll
  for (int i = 0; i < 8; ++i) { mrow[i] = -3.0e38f; lrow[i] = 0.0f; }

  __bf16* Pw = &plds[wv][0];

  // stage tile 0
  v8bf kreg = *(const v8bf*)(Kb + tid * 8);                  // K rows contiguous
  v8bf vreg = *(const v8bf*)(Vb + (size_t)vrow * SEQ + voff);
  *(v8bf*)(&kt[0][tid * 8]) = kreg;
  *(v8bf*)(&vt[0][vrow * 32 + voff]) = vreg;
  __syncthreads();

  for (int jt = 0; jt < SEQ / 32; ++jt) {
    const int j0  = jt * 32;
    const int buf = jt & 1;
    const bool more = (jt + 1) < SEQ / 32;
    if (more) {  // issue next tile's global loads; land after compute
      kreg = *(const v8bf*)(Kb + (size_t)(j0 + 32) * HDIM + tid * 8);
      vreg = *(const v8bf*)(Vb + (size_t)vrow * SEQ + (j0 + 32) + voff);
    }

    // S tiles from LDS K: B column = local key (sub*16+row), contiguous d.
    v16bf bk00 = b_frag_bf16(&kt[buf][(0  + row) * HDIM + 0  + hi * 16]);
    v16bf bk01 = b_frag_bf16(&kt[buf][(0  + row) * HDIM + 32 + hi * 16]);
    v16bf bk10 = b_frag_bf16(&kt[buf][(16 + row) * HDIM + 0  + hi * 16]);
    v16bf bk11 = b_frag_bf16(&kt[buf][(16 + row) * HDIM + 32 + hi * 16]);

    v8f s0 = vzero8(), s1 = vzero8();
    s0 = wmma_bf16(aq0, bk00, s0);
    s0 = wmma_bf16(aq1, bk01, s0);
    s1 = wmma_bf16(aq0, bk10, s1);
    s1 = wmma_bf16(aq1, bk11, s1);

    // Online softmax per row (rows per-VGPR-slot; cols across 16 lanes).
#pragma unroll
    for (int i = 0; i < 8; ++i) {
      float v0 = s0[i], v1 = s1[i];
      float mx = fmaxf(v0, v1);
#pragma unroll
      for (int off = 1; off < 16; off <<= 1)
        mx = fmaxf(mx, __shfl_xor(mx, off, 32));
      const float mn = fmaxf(mrow[i], mx);
      const float sc = __expf(mrow[i] - mn);
      const float p0 = __expf(v0 - mn);
      const float p1 = __expf(v1 - mn);
      float rs = p0 + p1;
#pragma unroll
      for (int off = 1; off < 16; off <<= 1)
        rs += __shfl_xor(rs, off, 32);
      lrow[i] = lrow[i] * sc + rs;
      mrow[i] = mn;
#pragma unroll
      for (int t = 0; t < 4; ++t) o[t][i] *= sc;
      const int r = i + hi * 8;                // C-layout row for this slot
      Pw[r * 32 + row]      = (__bf16)p0;
      Pw[r * 32 + 16 + row] = (__bf16)p1;
    }

    // Re-read P in A-fragment layout (same-wave LDS ops are in order).
    v16bf ap = a_frag_bf16(Pw + row * 32 + hi * 16);

    // O += P(16x32) * V(32x64): B columns = d rows of V^T tile in LDS.
#pragma unroll
    for (int t = 0; t < 4; ++t) {
      v16bf bv = b_frag_bf16(&vt[buf][(t * 16 + row) * 32 + hi * 16]);
      o[t] = wmma_bf16(ap, bv, o[t]);
    }

    if (more) {  // write next tile into the other buffer
      *(v8bf*)(&kt[buf ^ 1][tid * 8]) = kreg;
      *(v8bf*)(&vt[buf ^ 1][vrow * 32 + voff]) = vreg;
    }
    __syncthreads();  // reads of buf done; writes to buf^1 visible
  }

  float inv[8];
#pragma unroll
  for (int i = 0; i < 8; ++i) inv[i] = 1.0f / lrow[i];

#pragma unroll
  for (int t = 0; t < 4; ++t) {
#pragma unroll
    for (int i = 0; i < 8; ++i) {
      const int n = q0 + hi * 8 + i;
      O[((size_t)b * SEQ + n) * EMBED + h * HDIM + t * 16 + row] =
          (__bf16)(o[t][i] * inv[i]);
    }
  }
}

// ---------------------------------------------------------------------------
// Kernel 3: out = O @ w_out^T + b_out (f32 result).
// ---------------------------------------------------------------------------
__global__ __launch_bounds__(256)
void out_proj_kernel(const __bf16* __restrict__ O, const __bf16* __restrict__ w,
                     const float* __restrict__ bias, float* __restrict__ out) {
  const int lane = threadIdx.x & 31;
  const int wv   = threadIdx.x >> 5;
  const int row  = lane & 15;
  const int hi   = lane >> 4;
  const int tileM = blockIdx.y * 64 + (wv & 3) * 16;
  const int tileN = blockIdx.x * 128 + (wv >> 2) * 64;

  const __bf16* arow = O + (size_t)(tileM + row) * EMBED + hi * 16;
  const __bf16* bcol[4];
#pragma unroll
  for (int t = 0; t < 4; ++t)
    bcol[t] = w + (size_t)(tileN + t * 16 + row) * EMBED + hi * 16;

  v8f acc[4];
#pragma unroll
  for (int t = 0; t < 4; ++t) acc[t] = vzero8();

  gemm_core_16x64(arow, bcol, acc);

#pragma unroll
  for (int t = 0; t < 4; ++t) {
#pragma unroll
    for (int i = 0; i < 8; ++i) {
      const int m   = tileM + hi * 8 + i;
      const int col = tileN + t * 16 + row;
      out[(size_t)m * EMBED + col] = acc[t][i] + bias[col];
    }
  }
}

// ---------------------------------------------------------------------------
extern "C" void kernel_launch(void* const* d_in, const int* in_sizes, int n_in,
                              void* d_out, int out_size, void* d_ws, size_t ws_size,
                              hipStream_t stream) {
  const float* x    = (const float*)d_in[0];
  const float* wqkv = (const float*)d_in[1];
  const float* bqkv = (const float*)d_in[2];
  const float* wout = (const float*)d_in[3];
  const float* bout = (const float*)d_in[4];
  float* out = (float*)d_out;

  __bf16* ws = (__bf16*)d_ws;
  const size_t NE   = (size_t)BATCH * NHEAD * SEQ * HDIM;  // 8,388,608 elems
  const size_t NWQ  = (size_t)F3 * EMBED;                  // 3,145,728
  const size_t NWO  = (size_t)EMBED * EMBED;               // 1,048,576
  __bf16* Qs  = ws;
  __bf16* Ksb = ws + NE;
  __bf16* Vtb = ws + 2 * NE;
  __bf16* Ob  = ws + 3 * NE;
  __bf16* Xb  = ws + 4 * NE;          // bf16 copy of x (NE elems)
  __bf16* Wqb = ws + 5 * NE;          // bf16 copy of w_qkv
  __bf16* Wob = Wqb + NWQ;            // bf16 copy of w_out  (~88 MB total)

  // Pre-convert operands to bf16 (elementwise, bandwidth-trivial).
  cvt_f32_bf16_kernel<<<dim3((unsigned)(NE  / 2048)), 256, 0, stream>>>(x, Xb, (int)NE);
  cvt_f32_bf16_kernel<<<dim3((unsigned)(NWQ / 2048)), 256, 0, stream>>>(wqkv, Wqb, (int)NWQ);
  cvt_f32_bf16_kernel<<<dim3((unsigned)(NWO / 2048)), 256, 0, stream>>>(wout, Wob, (int)NWO);

  // M=8192, N=3072: grid (3072/128, 8192/64)
  qkv_proj_kernel<<<dim3(F3 / 128, MROWS / 64), 256, 0, stream>>>(
      Xb, Wqb, bqkv, Qs, Ksb, Vtb);
  // 64 (b,h) pairs x 16 row-blocks of 128
  attn_kernel<<<dim3(SEQ / 128, BATCH * NHEAD), 256, 0, stream>>>(
      Qs, Ksb, Vtb, Ob);
  // M=8192, N=1024
  out_proj_kernel<<<dim3(EMBED / 128, MROWS / 64), 256, 0, stream>>>(
      Ob, Wob, bout, out);
}
